// ResizeBack_63797444215321
// MI455X (gfx1250) — compile-verified, matrix operand order checked by reading, not run on hardware
//
#include <hip/hip_runtime.h>
#include <hip/hip_bf16.h>
#include <stdint.h>

// ---------------------------------------------------------------------------
// Types for CDNA5 WMMA
// ---------------------------------------------------------------------------
typedef __attribute__((ext_vector_type(16))) __bf16 v16bf;
typedef __attribute__((ext_vector_type(8)))  float  v8f;
typedef __attribute__((ext_vector_type(4)))  float  v4f;

#define PI_F 3.14159265358979323846f
#define KS   21            // kernel size
#define TW   16            // tile width  (output cols per WG) == WMMA M
#define TR   64            // tile height (output rows per WG)
#define XR   (TR + KS - 1) // 84 padded-input rows needed
#define XC   (TW + KS - 1) // 36 padded-input cols needed
#define XP   48            // LDS row pitch (bf16); cols XC..47 zero-filled so
                           // A-operand loads need no per-element masking

// output layout (floats)
#define N_OUT   (16LL * 3 * 1024 * 1024)   // 50331648
#define K_OFF   (N_OUT)                    // kernel  16*441
#define R_OFF   (K_OFF + 7056)             // radius  16
#define S_OFF   (R_OFF + 16)               // is_sinc 16
#define O_OFF   (S_OFF + 16)               // omega   16
#define M_OFF   (O_OFF + 16)               // mode    48

// LDS partition for conv kernel (bytes)
#define X_BYTES (XR * XP * 2)              // 8064
#define W_OFFB  (X_BYTES)                  // 441 f32 = 1764 B
#define G_OFFB  (X_BYTES + 1776)           // 9840, 16B aligned
#define G_BYTES (XR * KS * 16 * 4)         // 112896
#define SMEM_BYTES (G_OFFB + G_BYTES)      // 122736

// ---------------------------------------------------------------------------
// JAX threefry2x32-20 (exact reproduction of jax.random for key(42))
// ---------------------------------------------------------------------------
__device__ __forceinline__ uint32_t rotl32(uint32_t x, int r) {
  return (x << r) | (x >> (32 - r));
}

__device__ void tf2x32(uint32_t k0, uint32_t k1, uint32_t c0, uint32_t c1,
                       uint32_t* o0, uint32_t* o1) {
  const int ra[4] = {13, 15, 26, 6};
  const int rb[4] = {17, 29, 16, 24};
  uint32_t ks0 = k0, ks1 = k1, ks2 = 0x1BD11BDAu ^ k0 ^ k1;
  uint32_t x0 = c0 + ks0, x1 = c1 + ks1;
#pragma unroll
  for (int q = 0; q < 4; ++q) { x0 += x1; x1 = rotl32(x1, ra[q]); x1 ^= x0; }
  x0 += ks1; x1 += ks2 + 1u;
#pragma unroll
  for (int q = 0; q < 4; ++q) { x0 += x1; x1 = rotl32(x1, rb[q]); x1 ^= x0; }
  x0 += ks2; x1 += ks0 + 2u;
#pragma unroll
  for (int q = 0; q < 4; ++q) { x0 += x1; x1 = rotl32(x1, ra[q]); x1 ^= x0; }
  x0 += ks0; x1 += ks1 + 3u;
#pragma unroll
  for (int q = 0; q < 4; ++q) { x0 += x1; x1 = rotl32(x1, rb[q]); x1 ^= x0; }
  x0 += ks1; x1 += ks2 + 4u;
#pragma unroll
  for (int q = 0; q < 4; ++q) { x0 += x1; x1 = rotl32(x1, ra[q]); x1 ^= x0; }
  x0 += ks2; x1 += ks0 + 5u;
  *o0 = x0; *o1 = x1;
}

// random_bits(key, 32, (16,)) element i: counts=iota(16) split in halves.
__device__ __forceinline__ uint32_t jax_bits16(uint32_t k0, uint32_t k1, int i) {
  uint32_t o0, o1;
  if (i < 8) { tf2x32(k0, k1, (uint32_t)i, (uint32_t)(i + 8), &o0, &o1); return o0; }
  tf2x32(k0, k1, (uint32_t)(i - 8), (uint32_t)i, &o0, &o1); return o1;
}

__device__ __forceinline__ float u01(uint32_t bits) {
  uint32_t u = (bits >> 9) | 0x3f800000u;
  return __uint_as_float(u) - 1.0f;
}

// Numerical-Recipes J1, float32 (matches reference)
__device__ float bessel_j1f(float x) {
  float ax = fabsf(x);
  float y = x * x;
  float p1 = x * (72362614232.0f + y * (-7895059235.0f + y * (242396853.1f +
             y * (-2972611.439f + y * (15704.4826f + y * (-30.16036606f))))));
  float q1 = 144725228442.0f + y * (2300535178.0f + y * (18583304.74f +
             y * (99447.43394f + y * (376.9991397f + y))));
  float smallv = p1 / q1;
  float axs = fmaxf(ax, 1e-20f);
  float z = 8.0f / axs;
  float y2 = z * z;
  float xx = ax - 2.356194491f;
  float p2 = 1.0f + y2 * (0.00183105f + y2 * (-3.516396496e-05f +
             y2 * (2.457520174e-06f + y2 * (-2.40337019e-07f))));
  float q2 = 0.04687499995f + y2 * (-0.0002002690873f + y2 * (8.449199096e-06f +
             y2 * (-8.8228987e-07f + y2 * (1.05787412e-07f))));
  float sgn = (x > 0.0f) ? 1.0f : ((x < 0.0f) ? -1.0f : 0.0f);
  float bigv = sqrtf(0.636619772f / axs) * (cosf(xx) * p2 - z * sinf(xx) * q2) * sgn;
  return (ax < 8.0f) ? smallv : bigv;
}

// ---------------------------------------------------------------------------
// Kernel 1: generate the 16 sinc kernels + aux outputs (exact JAX PRNG replay)
// ---------------------------------------------------------------------------
__global__ __launch_bounds__(512) void gen_kernels(float* __restrict__ out,
                                                   float* __restrict__ wbuf) {
  __shared__ float vals[512];
  __shared__ float s_omega, s_r, s_issinc;
  const int b = blockIdx.x;
  const int t = threadIdx.x;

  if (t == 0) {
    // key = jax.random.key(42) -> data (0, 42); split into 3 keys:
    // counts iota(6), halves (0,1,2)/(3,4,5); reshape(3,2) of concat(y0,y1).
    uint32_t a0, a1, b0, b1, c0, c1;
    tf2x32(0u, 42u, 0u, 3u, &a0, &a1);
    tf2x32(0u, 42u, 1u, 4u, &b0, &b1);
    tf2x32(0u, 42u, 2u, 5u, &c0, &c1);
    const uint32_t kr0 = a0, kr1 = b0;   // k_r
    const uint32_t ko0 = c0, ko1 = a1;   // k_om
    const uint32_t ks0 = b1, ks1 = c1;   // k_sinc
    uint32_t rb = jax_bits16(kr0, kr1, b);
    uint32_t ob = jax_bits16(ko0, ko1, b);
    uint32_t sb = jax_bits16(ks0, ks1, b);
    int radius = 3 + (int)(rb % 9u);                 // randint(3, 12)
    float ratio = u01(ob);
    float issinc = (u01(sb) < 1.0f) ? 1.0f : 0.0f;   // SINC_PROB = 1.0
    int sinc_size = radius * 2 + 1;
    float minw = (sinc_size < 13) ? (PI_F / 3.0f) : (PI_F / 5.0f);
    float omega = ratio * (PI_F - minw) + minw;
    s_omega = omega; s_r = (float)radius; s_issinc = issinc;
  }
  __syncthreads();
  const float omega = s_omega, r = s_r, issinc = s_issinc;

  float val = 0.0f;
  if (t < KS * KS) {
    int i = t / KS, j = t - (t / KS) * KS;
    float xx = (float)(i - KS / 2), yy = (float)(j - KS / 2);
    float dist = xx * xx + yy * yy;          // squared distance (per reference)
    float dsafe = (dist == 0.0f) ? 1.0f : dist;
    float sk;
    if (i == KS / 2 && j == KS / 2) sk = omega * omega / (4.0f * PI_F);
    else sk = omega * bessel_j1f(omega * dist) / (2.0f * PI_F * dsafe);
    float pulse = (i == KS / 2 && j == KS / 2) ? 1.0f : 0.0f;
    float kv = sk * issinc + pulse * (1.0f - issinc);
    float mask = ((fabsf(xx) - r) <= 0.0f && (fabsf(yy) - r) <= 0.0f) ? 1.0f : 0.0f;
    val = kv * mask;
  }
  vals[t] = val;
  __syncthreads();
  for (int s = 256; s > 0; s >>= 1) {
    if (t < s) vals[t] += vals[t + s];
    __syncthreads();
  }
  const float sum = vals[0];
  const float nv = val / sum;
  if (t < KS * KS) {
    wbuf[b * (KS * KS) + t] = nv;                    // weights for conv stage
    out[K_OFF + (size_t)b * (KS * KS) + t] = nv;     // kernel output
  }
  if (t == 0) {
    out[R_OFF + b] = r;
    out[S_OFF + b] = issinc;
    out[O_OFF + b] = omega * issinc;
  }
  if (b == 0 && t < 48) out[M_OFF + t] = ((t % 3) == 2) ? 1.0f : 0.0f;
}

// ---------------------------------------------------------------------------
// Kernel 2: fused bilinear-2x upsample + reflect pad + depthwise 21x21 conv.
// Stage1 as Toeplitz-GEMM on v_wmma_f32_16x16x32_bf16, Stage2 diagonal add.
// ---------------------------------------------------------------------------
__device__ __forceinline__ int refl1024(int t) {
  if (t < 0) t = -t;
  if (t > 1023) t = 2046 - t;
  return t;
}
// half-pixel bilinear 2x weights with edge renormalization (== index clamp)
__device__ __forceinline__ void up2w(int u, int* i0, int* i1, float* w0, float* w1) {
  int t = u >> 1;
  int a, bb; float wa, wb;
  if ((u & 1) == 0) { a = t - 1; bb = t;     wa = 0.25f; wb = 0.75f; }
  else              { a = t;     bb = t + 1; wa = 0.75f; wb = 0.25f; }
  a = min(511, max(0, a)); bb = min(511, max(0, bb));
  *i0 = a; *i1 = bb; *w0 = wa; *w1 = wb;
}

__global__ __launch_bounds__(256) void conv_tiles(const float* __restrict__ x,
                                                  const float* __restrict__ wbuf,
                                                  float* __restrict__ out) {
  extern __shared__ char smem[];
  __bf16* Xs = (__bf16*)smem;                 // [XR][XP] padded-upsampled tile
  float*  Wsh = (float*)(smem + W_OFFB);      // [21*21] weights
  float*  Gs  = (float*)(smem + G_OFFB);      // [XR][21][16] row-conv results

  const int id  = blockIdx.x;
  const int tx  = id & 63;                    // 64 column tiles
  const int ty  = (id >> 6) & 15;             // 16 row tiles
  const int img = id >> 10;                   // 48 = 16 b * 3 ch
  const int bb  = img / 3;
  const int r0  = ty * TR, c0 = tx * TW;
  const int tid = threadIdx.x;
  const float* S = x + (size_t)img * (512 * 512);

  // ---- Stage 0: upsample+reflect-pad tile on the fly -> LDS (bf16) ----
  // Fill the whole pitch (XP); cols >= XC are zeroed so stage-1 A-operand
  // reads are unconditional and can never feed NaN bit-patterns into WMMA.
  for (int e = tid; e < XR * XP; e += 256) {
    int yy = e / XP, p = e - yy * XP;
    float val = 0.0f;
    if (p < XC) {
      int u = refl1024(r0 + yy - (KS / 2));
      int v = refl1024(c0 + p  - (KS / 2));
      int ri0, ri1, ci0, ci1; float wr0, wr1, wc0, wc1;
      up2w(u, &ri0, &ri1, &wr0, &wr1);
      up2w(v, &ci0, &ci1, &wc0, &wc1);
      val = wr0 * (wc0 * S[ri0 * 512 + ci0] + wc1 * S[ri0 * 512 + ci1]) +
            wr1 * (wc0 * S[ri1 * 512 + ci0] + wc1 * S[ri1 * 512 + ci1]);
    }
    Xs[yy * XP + p] = (__bf16)val;
  }
  for (int e = tid; e < KS * KS; e += 256) Wsh[e] = wbuf[bb * (KS * KS) + e];
  __syncthreads();

  const int lane = tid & 31;
  const int wave = tid >> 5;
  const int cn   = lane & 15;      // M row (output col) / N col (kernel row)
  const int hi   = lane >> 4;
  const int jlo  = hi ? 8 : 0;     // 16-bit A/B layout: lower lanes K 0-7,16-23
  const int jhi  = hi ? 24 : 16;   //                    upper lanes K 8-15,24-31

  // ---- B matrices (kernel^T): B[k=j][n=i] = w[i][j]; built once.
  // K rows j in [21,32) are exactly 0.0 -> A garbage there contributes 0.
  v16bf B0, B1;
  const int i0c = cn, i1c = 16 + cn;
#pragma unroll
  for (int q = 0; q < 8; ++q) {
    int j = jlo + q;
    B0[q]     = (__bf16)((j < KS) ? Wsh[i0c * KS + j] : 0.0f);
    B1[q]     = (__bf16)((i1c < KS && j < KS) ? Wsh[i1c * KS + j] : 0.0f);
    j = jhi + q;
    B0[8 + q] = (__bf16)((j < KS) ? Wsh[i0c * KS + j] : 0.0f);
    B1[8 + q] = (__bf16)((i1c < KS && j < KS) ? Wsh[i1c * KS + j] : 0.0f);
  }

  // ---- Stage 1: per padded row y, G[y][i][c] = sum_j w[i][j]*X[y][c+j] ----
  for (int yy = wave; yy < XR; yy += 8) {        // wave-uniform -> EXEC all 1s
    const __bf16* xr = Xs + yy * XP + cn;        // A[c][j] = Toeplitz window
    v16bf A;
#pragma unroll
    for (int q = 0; q < 8; ++q) {                // unconditional: c+31 <= 46 < XP
      A[q]     = xr[jlo + q];
      A[8 + q] = xr[jhi + q];
    }
    v8f cz = {};
    v8f d0 = __builtin_amdgcn_wmma_f32_16x16x32_bf16(false, A, false, B0,
                                                     (short)0, cz, false, false);
    v8f d1 = __builtin_amdgcn_wmma_f32_16x16x32_bf16(false, A, false, B1,
                                                     (short)0, cz, false, false);
    // D layout: vgpr v, lanes0-15 -> M=v, lanes16-31 -> M=v+8; N = lane%16.
    // 8 contiguous floats per lane, 32B-aligned -> two ds_store_b128.
    const int chalf = hi * 8;
    {
      v4f* g = (v4f*)(Gs + ((yy * KS + cn) * 16 + chalf));
      v4f lo = {d0[0], d0[1], d0[2], d0[3]};
      v4f hi4 = {d0[4], d0[5], d0[6], d0[7]};
      g[0] = lo; g[1] = hi4;
    }
    const int iB = 16 + cn;
    if (iB < KS) {
      v4f* g = (v4f*)(Gs + ((yy * KS + iB) * 16 + chalf));
      v4f lo = {d1[0], d1[1], d1[2], d1[3]};
      v4f hi4 = {d1[4], d1[5], d1[6], d1[7]};
      g[0] = lo; g[1] = hi4;
    }
  }
  __syncthreads();

  // ---- Stage 2: out[r][c] = sum_i G[r+i][i][c]; conflict-free banks ----
  float* O = out + (size_t)img * (1024 * 1024);
  for (int o = tid; o < TR * TW; o += 256) {
    int rr = o >> 4, cc = o & 15;
    float acc = 0.0f;
#pragma unroll
    for (int i = 0; i < KS; ++i) acc += Gs[((rr + i) * KS + i) * 16 + cc];
    O[(size_t)(r0 + rr) * 1024 + (c0 + cc)] = acc;
  }
}

// ---------------------------------------------------------------------------
extern "C" void kernel_launch(void* const* d_in, const int* in_sizes, int n_in,
                              void* d_out, int out_size, void* d_ws, size_t ws_size,
                              hipStream_t stream) {
  (void)in_sizes; (void)n_in; (void)out_size; (void)ws_size;
  const float* x = (const float*)d_in[0];   // (16,3,512,512) f32
  float* out = (float*)d_out;
  float* wbuf = (float*)d_ws;               // 16*441 f32 weights scratch

  // allow the 122 KB dynamic LDS footprint (no-op if already permitted)
  (void)hipFuncSetAttribute((const void*)conv_tiles,
                            hipFuncAttributeMaxDynamicSharedMemorySize,
                            SMEM_BYTES);

  gen_kernels<<<16, 512, 0, stream>>>(out, wbuf);
  conv_tiles<<<dim3(64 * 16 * 48), dim3(256), SMEM_BYTES, stream>>>(x, wbuf, out);
}
